// AEDCBlock_50294067036289
// MI455X (gfx1250) — matrix operand order repaired and checked by reading, not compiled, not run on hardware
//
#include <hip/hip_runtime.h>

// ---------------------------------------------------------------------------
// MI455X / gfx1250 implementation. wave32, WMMA f32<-f16 16x16x32 everywhere
// the GEMMs are hot; flash-style streaming softmax so the 4x(4096x4096)
// attention matrices are never materialized (saves ~1GB of HBM traffic).
// ---------------------------------------------------------------------------

typedef __attribute__((ext_vector_type(16))) _Float16 v16h;
typedef __attribute__((ext_vector_type(8)))  _Float16 v8h;
typedef __attribute__((ext_vector_type(8)))  float    v8f;

#define BATCH   2
#define HWSZ    4096      // 64*64
#define CCH     64
#define NB      4
#define CAT     256
#define NPOOL   4
#define FQ      16
#define CFUSED  320
#define BNEPS   1e-3f

__device__ __forceinline__ v8f wmma16(v16h a, v16h b, v8f c) {
  return __builtin_amdgcn_wmma_f32_16x16x32_f16(false, a, false, b,
                                                (short)0, c, false, false);
}
__device__ __forceinline__ v16h zro16() {
  v16h a;
#pragma unroll
  for (int j = 0; j < 16; ++j) a[j] = (_Float16)0.f;
  return a;
}

// ---------------------------------------------------------------------------
// Prep kernels: fp32 -> f16 conversions and weight transposes so every WMMA
// B-tile is two 16B per-lane loads from one co row.
// ---------------------------------------------------------------------------
__global__ void cvt_x_k(const float* __restrict__ x, _Float16* __restrict__ xh, int n) {
  int i = blockIdx.x * blockDim.x + threadIdx.x;
  if (i < n) xh[i] = (_Float16)x[i];
}

// conv_w [NB][3][3][ci=64][co=64] -> wth [NB][9][co][ci] (f16)
__global__ void cvt_wconv_k(const float* __restrict__ w, _Float16* __restrict__ wth) {
  int idx = blockIdx.x * blockDim.x + threadIdx.x;
  if (idx >= NB * 9 * 64 * 64) return;
  int ci  = idx & 63;
  int co  = (idx >> 6) & 63;
  int tap = (idx >> 12) % 9;
  int i   = idx / (9 * 4096);
  wth[idx] = (_Float16)w[(((size_t)i * 9 + tap) * 64 + ci) * 64 + co];
}

// v_w [NB][ci=64][co=64] -> vwt [NB][co][ci] (f16)
__global__ void cvt_vwt_k(const float* __restrict__ w, _Float16* __restrict__ vwt) {
  int idx = blockIdx.x * blockDim.x + threadIdx.x;
  if (idx >= NB * 64 * 64) return;
  int ci = idx & 63;
  int co = (idx >> 6) & 63;
  int i  = idx >> 12;
  vwt[idx] = (_Float16)w[((size_t)i * 64 + ci) * 64 + co];
}

// proj_w [ci=320][co=64] -> projwt [co][ci] (f16)
__global__ void cvt_projwt_k(const float* __restrict__ w, _Float16* __restrict__ wt) {
  int idx = blockIdx.x * blockDim.x + threadIdx.x;
  if (idx >= 64 * 320) return;
  int ci = idx % 320;
  int co = idx / 320;
  wt[idx] = (_Float16)w[(size_t)ci * 64 + co];
}

// ---------------------------------------------------------------------------
// Branch conv 3x3 dilated + BN + ReLU  (implicit im2col WMMA GEMM)
// grid = (128, NB), block = 128 (4 waves); wave tile: M=16 pixels x N=64 ch.
// ---------------------------------------------------------------------------
__global__ __launch_bounds__(128)
void conv_bn_relu_k(const _Float16* __restrict__ xh,
                    const _Float16* __restrict__ wth,
                    const float* __restrict__ conv_b, const float* __restrict__ bn_g,
                    const float* __restrict__ bn_b,   const float* __restrict__ bn_m,
                    const float* __restrict__ bn_v,
                    float* __restrict__ bf, _Float16* __restrict__ bh) {
  const int i    = blockIdx.y;
  const int wave = threadIdx.x >> 5;
  const int lane = threadIdx.x & 31;
  const int tile = blockIdx.x * 4 + wave;     // [0,512)
  const int P0   = tile * 16;                 // global pixel over B*HW
  const int bb   = P0 >> 12;
  const int p    = P0 & 4095;
  const int hh   = p >> 6;
  const int ww0  = p & 63;
  const int dil  = 1 << i;                    // 1,2,4,8
  const int r    = lane & 15;
  const int half = lane >> 4;
  const int kb   = half ? 8 : 0;

  v8f acc[4] = {};
  const _Float16* wbase = wth + (size_t)i * 9 * 64 * 64;
#pragma unroll
  for (int tap = 0; tap < 9; ++tap) {
    const int dy = tap / 3 - 1, dx = tap % 3 - 1;
    const int hs = hh + dy * dil;
    const int ws = ww0 + r + dx * dil;
    const bool valid = (hs >= 0) & (hs < 64) & (ws >= 0) & (ws < 64);
    const _Float16* wtap = wbase + tap * 64 * 64;
#pragma unroll
    for (int kc = 0; kc < 2; ++kc) {
      v16h a = zro16();
      if (valid) {
        const _Float16* xrow =
            xh + (((size_t)bb * HWSZ + hs * 64 + ws) << 6) + kc * 32 + kb;
        union { v16h v; v8h h[2]; } ua;
        ua.h[0] = *(const v8h*)(xrow);
        ua.h[1] = *(const v8h*)(xrow + 16);
        a = ua.v;
      }
#pragma unroll
      for (int n = 0; n < 4; ++n) {
        const _Float16* wrow = wtap + (n * 16 + r) * 64 + kc * 32 + kb;
        union { v16h v; v8h h[2]; } ub;
        ub.h[0] = *(const v8h*)(wrow);
        ub.h[1] = *(const v8h*)(wrow + 16);
        acc[n] = wmma16(a, ub.v, acc[n]);
      }
    }
  }
#pragma unroll
  for (int n = 0; n < 4; ++n) {
    const int co = n * 16 + r;
    const float bias = conv_b[i * 64 + co];
    const float sc   = bn_g[i * 64 + co] * rsqrtf(bn_v[i * 64 + co] + BNEPS);
    const float mu   = bn_m[i * 64 + co];
    const float be   = bn_b[i * 64 + co];
#pragma unroll
    for (int v = 0; v < 8; ++v) {
      const int m = v + half * 8;
      float y = (acc[n][v] + bias - mu) * sc + be;
      y = fmaxf(y, 0.f);
      const size_t idx = (((size_t)i * BATCH + bb) * HWSZ + p + m) * CCH + co;
      bf[idx] = y;
      bh[idx] = (_Float16)y;
    }
  }
}

// ---------------------------------------------------------------------------
// Q/K projections (tiny: 67 MFLOP total) — scalar VALU, output f16.
// ---------------------------------------------------------------------------
__global__ void qk_gemm_k(const _Float16* __restrict__ bh,
                          const float* __restrict__ q_w, const float* __restrict__ q_b,
                          const float* __restrict__ k_w, const float* __restrict__ k_b,
                          _Float16* __restrict__ qh, _Float16* __restrict__ kh) {
  int idx = blockIdx.x * blockDim.x + threadIdx.x;        // NB*B*HW*8
  if (idx >= NB * BATCH * HWSZ * 8) return;
  const int d  = idx & 7;
  const int pg = idx >> 3;
  const int i  = pg >> 13;
  const int bp = pg & 8191;
  const _Float16* xrow = bh + ((size_t)i * BATCH * HWSZ + bp) * CCH;
  const float* qw = q_w + (size_t)i * 64 * 8;
  const float* kw = k_w + (size_t)i * 64 * 8;
  float q = q_b[i * 8 + d], k = k_b[i * 8 + d];
  for (int c = 0; c < 64; ++c) {
    const float xv = (float)xrow[c];
    q += xv * qw[c * 8 + d];
    k += xv * kw[c * 8 + d];
  }
  qh[idx] = (_Float16)q;
  kh[idx] = (_Float16)k;
}

// ---------------------------------------------------------------------------
// V projection: WMMA GEMM M=8192,N=64,K=64; writes V transposed f16 [i][b][co][p]
// so the flash kernel's B-tiles are contiguous.  grid=512, block=128.
// ---------------------------------------------------------------------------
__global__ __launch_bounds__(128)
void v_gemm_k(const _Float16* __restrict__ bh, const _Float16* __restrict__ vwt,
              const float* __restrict__ v_b, _Float16* __restrict__ vt) {
  const int wave = threadIdx.x >> 5;
  const int lane = threadIdx.x & 31;
  const int tile = blockIdx.x * 4 + wave;   // [0,2048)
  const int i    = tile >> 9;
  const int t    = tile & 511;
  const int P0   = t * 16;
  const int bb   = P0 >> 12;
  const int pp   = P0 & 4095;
  const int r    = lane & 15;
  const int half = lane >> 4;
  const int kb   = half ? 8 : 0;

  v8f acc[4] = {};
  const _Float16* arow0 = bh + ((size_t)i * BATCH * HWSZ + P0) * CCH;
  const _Float16* wbase = vwt + (size_t)i * 64 * 64;
#pragma unroll
  for (int kc = 0; kc < 2; ++kc) {
    union { v16h v; v8h h[2]; } ua;
    const _Float16* xrow = arow0 + (size_t)r * CCH + kc * 32 + kb;
    ua.h[0] = *(const v8h*)(xrow);
    ua.h[1] = *(const v8h*)(xrow + 16);
#pragma unroll
    for (int n = 0; n < 4; ++n) {
      const _Float16* wrow = wbase + (n * 16 + r) * 64 + kc * 32 + kb;
      union { v16h v; v8h h[2]; } ub;
      ub.h[0] = *(const v8h*)(wrow);
      ub.h[1] = *(const v8h*)(wrow + 16);
      acc[n] = wmma16(ua.v, ub.v, acc[n]);
    }
  }
#pragma unroll
  for (int n = 0; n < 4; ++n) {
    const int co = n * 16 + r;
    const float bias = v_b[i * 64 + co];
#pragma unroll
    for (int v = 0; v < 8; ++v) {
      const int m = v + half * 8;
      vt[((((size_t)i * BATCH + bb) * CCH) + co) * HWSZ + pp + m] =
          (_Float16)(acc[n][v] + bias);
    }
  }
}

// ---------------------------------------------------------------------------
// Flash attention per branch: 16 queries/wave, stream 4096 keys in chunks of
// 32.  S via f16 WMMA (dk=8 zero-padded), online softmax via width-16
// shuffles, P staged through wave-private LDS, P*V via 4 WMMAs against
// transposed V.  Output: gamma*O + branch, written into fused f16 buffer.
// grid = 512, block = 128.
// ---------------------------------------------------------------------------
__global__ __launch_bounds__(128)
void flash_attn_k(const _Float16* __restrict__ qh, const _Float16* __restrict__ kh,
                  const _Float16* __restrict__ vt, const float* __restrict__ bf,
                  const float* __restrict__ attn_gamma,
                  _Float16* __restrict__ fusedh) {
  __shared__ __align__(16) _Float16 ldsP[4][16][32];
  const int wave = threadIdx.x >> 5;
  const int lane = threadIdx.x & 31;
  const int tile = blockIdx.x * 4 + wave;   // [0,2048)
  const int i    = tile >> 9;
  const int t    = tile & 511;
  const int bb   = t >> 8;
  const int q0   = (t & 255) * 16;
  const int r    = lane & 15;
  const int half = lane >> 4;

  // Q tile in A layout (K padded 8 -> 32 with zeros; lanes>=16 all zero)
  v16h aq = zro16();
  if (half == 0) {
    const v8h qv = *(const v8h*)(qh + (((size_t)i * BATCH + bb) * HWSZ + q0 + r) * 8);
#pragma unroll
    for (int j = 0; j < 8; ++j) aq[j] = qv[j];
  }

  const _Float16* kbp = kh + (((size_t)i * BATCH + bb) * HWSZ) * 8;
  const _Float16* vbp = vt + (((size_t)i * BATCH + bb) * CCH) * HWSZ;

  v8f   O[4] = {};
  float mrun[8], lrun[8];
#pragma unroll
  for (int v = 0; v < 8; ++v) { mrun[v] = -1e30f; lrun[v] = 0.f; }

  for (int kv = 0; kv < HWSZ; kv += 32) {
    // prefetch next K/V chunk while this one computes
    if (kv + 32 < HWSZ) {
      __builtin_prefetch(kbp + (kv + 32 + r) * 8, 0, 3);
      __builtin_prefetch(vbp + (size_t)r * HWSZ + kv + 32, 0, 3);
    }
    // ---- S = Q * K^T, two 16x16 tiles ----
    v8f S[2];
#pragma unroll
    for (int s = 0; s < 2; ++s) {
      v16h bk = zro16();
      if (half == 0) {
        const v8h kvv = *(const v8h*)(kbp + (size_t)(kv + s * 16 + r) * 8);
#pragma unroll
        for (int j = 0; j < 8; ++j) bk[j] = kvv[j];
      }
      v8f z = {};
      S[s] = wmma16(aq, bk, z);
    }
    // ---- online softmax (row m = v + 8*half: lives across one 16-lane half) ----
#pragma unroll
    for (int v = 0; v < 8; ++v) {
      float mx = fmaxf(S[0][v], S[1][v]);
#pragma unroll
      for (int off = 1; off < 16; off <<= 1) mx = fmaxf(mx, __shfl_xor(mx, off, 16));
      const float mnew = fmaxf(mrun[v], mx);
      const float sc   = __expf(mrun[v] - mnew);
      const float p0   = __expf(S[0][v] - mnew);
      const float p1   = __expf(S[1][v] - mnew);
      float rs = p0 + p1;
#pragma unroll
      for (int off = 1; off < 16; off <<= 1) rs += __shfl_xor(rs, off, 16);
      lrun[v] = lrun[v] * sc + rs;
      mrun[v] = mnew;
#pragma unroll
      for (int n = 0; n < 4; ++n) O[n][v] *= sc;
      const int m = v + half * 8;
      ldsP[wave][m][r]      = (_Float16)p0;
      ldsP[wave][m][16 + r] = (_Float16)p1;
    }
    // wave-local LDS RAW: LDS is in-order per wave, just drain DScnt
    asm volatile("s_wait_dscnt 0x0" ::: "memory");
    // ---- P in A layout from LDS ----
    const int kb = half ? 8 : 0;
    union { v16h v; v8h h[2]; } up;
    up.h[0] = *(const v8h*)&ldsP[wave][r][kb];
    up.h[1] = *(const v8h*)&ldsP[wave][r][kb + 16];
    // ---- O += P * V ----
#pragma unroll
    for (int n = 0; n < 4; ++n) {
      const _Float16* vrow = vbp + (size_t)(n * 16 + r) * HWSZ + kv + kb;
      union { v16h v; v8h h[2]; } uv;
      uv.h[0] = *(const v8h*)(vrow);
      uv.h[1] = *(const v8h*)(vrow + 16);
      O[n] = wmma16(up.v, uv.v, O[n]);
    }
  }
  // ---- epilogue: out = gamma * softmax(QK)V + branch ----
  const float gamma = attn_gamma[i];
#pragma unroll
  for (int n = 0; n < 4; ++n) {
    const int co = n * 16 + r;
#pragma unroll
    for (int v = 0; v < 8; ++v) {
      const int m = v + half * 8;
      const float o = O[n][v] / lrun[v];
      const float y = gamma * o +
          bf[(((size_t)i * BATCH + bb) * HWSZ + q0 + m) * CCH + co];
      fusedh[((size_t)bb * HWSZ + q0 + m) * CFUSED + i * CCH + co] = (_Float16)y;
    }
  }
}

// ---------------------------------------------------------------------------
// Pyramid pooling path (<1% of FLOPs) — scalar VALU.
// ---------------------------------------------------------------------------
__global__ void avgpool_k(const _Float16* __restrict__ fusedh,
                          float* __restrict__ poolbuf, int ps, int pd) {
  int idx = blockIdx.x * blockDim.x + threadIdx.x;
  if (idx >= BATCH * pd * pd * CAT) return;
  const int c  = idx & 255;
  int rest = idx >> 8;
  const int pw = rest % pd; rest /= pd;
  const int ph = rest % pd;
  const int bb = rest / pd;
  float s = 0.f; int cnt = 0;
  for (int dy = 0; dy < ps; ++dy) {
    const int hs = ph * ps + dy;
    if (hs >= 64) break;
    for (int dx = 0; dx < ps; ++dx) {
      const int wsr = pw * ps + dx;
      if (wsr >= 64) break;
      s += (float)fusedh[((size_t)bb * HWSZ + hs * 64 + wsr) * CFUSED + c];
      ++cnt;
    }
  }
  poolbuf[idx] = s / (float)cnt;   // SAME avg-pool excludes padding
}

__global__ void ppl_conv_k(const float* __restrict__ poolbuf,
                           const _Float16* __restrict__ fusedh, int use_full,
                           int pd, int j,
                           const float* __restrict__ ppl_w, const float* __restrict__ ppl_b,
                           const float* __restrict__ g,  const float* __restrict__ be,
                           const float* __restrict__ mu, const float* __restrict__ va,
                           float* __restrict__ convbuf) {
  int idx = blockIdx.x * blockDim.x + threadIdx.x;
  if (idx >= BATCH * pd * pd * FQ) return;
  const int co = idx & 15;
  int rest = idx >> 4;
  const int pw = rest % pd; rest /= pd;
  const int ph = rest % pd;
  const int bb = rest / pd;
  const float* wj = ppl_w + (size_t)j * CAT * FQ;
  float s = ppl_b[j * FQ + co];
  if (use_full) {
    const _Float16* row = fusedh + ((size_t)bb * HWSZ + ph * 64 + pw) * CFUSED;
    for (int ci = 0; ci < CAT; ++ci) s += (float)row[ci] * wj[ci * FQ + co];
  } else {
    const float* row = poolbuf + (((size_t)bb * pd + ph) * pd + pw) * CAT;
    for (int ci = 0; ci < CAT; ++ci) s += row[ci] * wj[ci * FQ + co];
  }
  const float sc = g[j * FQ + co] * rsqrtf(va[j * FQ + co] + BNEPS);
  convbuf[idx] = (s - mu[j * FQ + co]) * sc + be[j * FQ + co];
}

__global__ void resize_k(const float* __restrict__ convbuf, int pd, int j,
                         _Float16* __restrict__ fusedh) {
  int idx = blockIdx.x * blockDim.x + threadIdx.x;
  if (idx >= BATCH * HWSZ * FQ) return;
  const int co = idx & 15;
  const int p  = (idx >> 4) & 4095;
  const int bb = idx >> 16;
  const int hh = p >> 6, ww = p & 63;
  const float scale = (float)pd / 64.f;
  float sy = ((float)hh + 0.5f) * scale - 0.5f;
  float sx = ((float)ww + 0.5f) * scale - 0.5f;
  sy = fminf(fmaxf(sy, 0.f), (float)(pd - 1));
  sx = fminf(fmaxf(sx, 0.f), (float)(pd - 1));
  const int y0 = (int)sy, x0 = (int)sx;
  const int y1 = (y0 + 1 < pd) ? y0 + 1 : pd - 1;
  const int x1 = (x0 + 1 < pd) ? x0 + 1 : pd - 1;
  const float fy = sy - (float)y0, fx = sx - (float)x0;
  const float* base = convbuf + (size_t)bb * pd * pd * FQ;
  const float v00 = base[(y0 * pd + x0) * FQ + co];
  const float v01 = base[(y0 * pd + x1) * FQ + co];
  const float v10 = base[(y1 * pd + x0) * FQ + co];
  const float v11 = base[(y1 * pd + x1) * FQ + co];
  const float val = v00 * (1 - fy) * (1 - fx) + v01 * (1 - fy) * fx +
                    v10 * fy * (1 - fx) + v11 * fy * fx;
  fusedh[((size_t)bb * HWSZ + p) * CFUSED + CAT + j * FQ + co] = (_Float16)val;
}

// ---------------------------------------------------------------------------
// Final projection 320->64 + BN + ReLU:  WMMA GEMM M=8192,N=64,K=320.
// grid = 128, block = 128.
// ---------------------------------------------------------------------------
__global__ __launch_bounds__(128)
void proj_k(const _Float16* __restrict__ fusedh, const _Float16* __restrict__ projwt,
            const float* __restrict__ proj_b,
            const float* __restrict__ g,  const float* __restrict__ be,
            const float* __restrict__ mu, const float* __restrict__ va,
            float* __restrict__ out) {
  const int wave = threadIdx.x >> 5;
  const int lane = threadIdx.x & 31;
  const int tile = blockIdx.x * 4 + wave;   // [0,512)
  const int P0   = tile * 16;
  const int r    = lane & 15;
  const int half = lane >> 4;
  const int kb   = half ? 8 : 0;

  v8f acc[4] = {};
  const _Float16* arow0 = fusedh + (size_t)P0 * CFUSED;
#pragma unroll
  for (int kc = 0; kc < 10; ++kc) {
    union { v16h v; v8h h[2]; } ua;
    const _Float16* xrow = arow0 + (size_t)r * CFUSED + kc * 32 + kb;
    ua.h[0] = *(const v8h*)(xrow);
    ua.h[1] = *(const v8h*)(xrow + 16);
#pragma unroll
    for (int n = 0; n < 4; ++n) {
      const _Float16* wrow = projwt + (size_t)(n * 16 + r) * CFUSED + kc * 32 + kb;
      union { v16h v; v8h h[2]; } ub;
      ub.h[0] = *(const v8h*)(wrow);
      ub.h[1] = *(const v8h*)(wrow + 16);
      acc[n] = wmma16(ua.v, ub.v, acc[n]);
    }
  }
#pragma unroll
  for (int n = 0; n < 4; ++n) {
    const int co = n * 16 + r;
    const float bias = proj_b[co];
    const float sc   = g[co] * rsqrtf(va[co] + BNEPS);
#pragma unroll
    for (int v = 0; v < 8; ++v) {
      const int m = v + half * 8;
      float y = (acc[n][v] + bias - mu[co]) * sc + be[co];
      out[(size_t)(P0 + m) * CCH + co] = fmaxf(y, 0.f);
    }
  }
}

// ---------------------------------------------------------------------------
extern "C" void kernel_launch(void* const* d_in, const int* in_sizes, int n_in,
                              void* d_out, int out_size, void* d_ws, size_t ws_size,
                              hipStream_t stream) {
  (void)in_sizes; (void)n_in; (void)out_size; (void)ws_size;
  const float* x      = (const float*)d_in[0];
  const float* conv_w = (const float*)d_in[1];
  const float* conv_b = (const float*)d_in[2];
  const float* bn_g   = (const float*)d_in[3];
  const float* bn_b   = (const float*)d_in[4];
  const float* bn_m   = (const float*)d_in[5];
  const float* bn_v   = (const float*)d_in[6];
  const float* q_w    = (const float*)d_in[7];
  const float* q_b    = (const float*)d_in[8];
  const float* k_w    = (const float*)d_in[9];
  const float* k_b    = (const float*)d_in[10];
  const float* v_w    = (const float*)d_in[11];
  const float* v_b    = (const float*)d_in[12];
  const float* gamma  = (const float*)d_in[13];
  const float* ppl_w  = (const float*)d_in[14];
  const float* ppl_b  = (const float*)d_in[15];
  const float* ppl_g  = (const float*)d_in[16];
  const float* ppl_be = (const float*)d_in[17];
  const float* ppl_m  = (const float*)d_in[18];
  const float* ppl_v  = (const float*)d_in[19];
  const float* proj_w = (const float*)d_in[20];
  const float* proj_b = (const float*)d_in[21];
  const float* pbn_g  = (const float*)d_in[22];
  const float* pbn_b  = (const float*)d_in[23];
  const float* pbn_m  = (const float*)d_in[24];
  const float* pbn_v  = (const float*)d_in[25];
  float* out = (float*)d_out;

  // workspace carve-up (~27 MB total)
  char* ws = (char*)d_ws;
  size_t off = 0;
  auto carve = [&](size_t bytes) -> void* {
    void* p = ws + off;
    off = (off + bytes + 255) & ~(size_t)255;
    return p;
  };
  _Float16* xh      = (_Float16*)carve((size_t)BATCH * HWSZ * CCH * 2);
  _Float16* wth     = (_Float16*)carve((size_t)NB * 9 * 64 * 64 * 2);
  _Float16* vwt     = (_Float16*)carve((size_t)NB * 64 * 64 * 2);
  _Float16* projwt  = (_Float16*)carve((size_t)64 * 320 * 2);
  float*    bf      = (float*)   carve((size_t)NB * BATCH * HWSZ * CCH * 4);
  _Float16* bh      = (_Float16*)carve((size_t)NB * BATCH * HWSZ * CCH * 2);
  _Float16* qh      = (_Float16*)carve((size_t)NB * BATCH * HWSZ * 8 * 2);
  _Float16* kh      = (_Float16*)carve((size_t)NB * BATCH * HWSZ * 8 * 2);
  _Float16* vt      = (_Float16*)carve((size_t)NB * BATCH * CCH * HWSZ * 2);
  _Float16* fusedh  = (_Float16*)carve((size_t)BATCH * HWSZ * CFUSED * 2);
  float*    poolbuf = (float*)   carve((size_t)BATCH * 32 * 32 * CAT * 4);
  float*    convbuf = (float*)   carve((size_t)BATCH * HWSZ * FQ * 4);

  // --- prep ---
  cvt_x_k     <<<2048, 256, 0, stream>>>(x, xh, BATCH * HWSZ * CCH);
  cvt_wconv_k <<<576,  256, 0, stream>>>(conv_w, wth);
  cvt_vwt_k   <<<64,   256, 0, stream>>>(v_w, vwt);
  cvt_projwt_k<<<80,   256, 0, stream>>>(proj_w, projwt);

  // --- branch convs (WMMA) ---
  conv_bn_relu_k<<<dim3(128, NB), 128, 0, stream>>>(
      xh, wth, conv_b, bn_g, bn_b, bn_m, bn_v, bf, bh);

  // --- q/k projections (scalar) and v projection (WMMA, transposed output) ---
  qk_gemm_k<<<1024, 256, 0, stream>>>(bh, q_w, q_b, k_w, k_b, qh, kh);
  v_gemm_k <<<512, 128, 0, stream>>>(bh, vwt, v_b, vt);

  // --- flash attention (WMMA) ---
  flash_attn_k<<<512, 128, 0, stream>>>(qh, kh, vt, bf, gamma, fusedh);

  // --- pyramid pooling ---
  const int PS[NPOOL] = {1, 2, 3, 6};
  const int PD[NPOOL] = {64, 32, 22, 11};
  for (int j = 0; j < NPOOL; ++j) {
    const int ps = PS[j], pd = PD[j];
    if (ps > 1) {
      const int n = BATCH * pd * pd * CAT;
      avgpool_k<<<(n + 255) / 256, 256, 0, stream>>>(fusedh, poolbuf, ps, pd);
    }
    const int nc = BATCH * pd * pd * FQ;
    ppl_conv_k<<<(nc + 255) / 256, 256, 0, stream>>>(
        poolbuf, fusedh, ps == 1 ? 1 : 0, pd, j,
        ppl_w, ppl_b, ppl_g, ppl_be, ppl_m, ppl_v, convbuf);
    const int nr = BATCH * HWSZ * FQ;
    resize_k<<<(nr + 255) / 256, 256, 0, stream>>>(convbuf, pd, j, fusedh);
  }

  // --- final projection (WMMA) ---
  proj_k<<<128, 128, 0, stream>>>(fusedh, projwt, proj_b,
                                  pbn_g, pbn_b, pbn_m, pbn_v, out);
}
